// MultiHeadSelfAttention_26980984554163
// MI455X (gfx1250) — matrix operand order repaired and checked by reading, not compiled, not run on hardware
//
#include <hip/hip_runtime.h>
#include <hip/hip_bf16.h>

// ---------------------------------------------------------------------------
// MHA forward for MI455X (gfx1250). bf16 WMMA (v_wmma_f32_16x16x32_bf16) with
// fp32 accumulation for all four GEMMs; exact two-pass online softmax over the
// mandatory 1.07 GB attn output (bandwidth floor ~100us at 23.3 TB/s).
// Weight tiles are staged into LDS with the Tensor Data Mover when the
// toolchain exposes __builtin_amdgcn_tensor_load_to_lds (guarded; manual
// cooperative staging otherwise).
// ---------------------------------------------------------------------------

typedef __attribute__((ext_vector_type(16))) __bf16          v16bf;
typedef __attribute__((ext_vector_type(8)))  float           v8f;
typedef __attribute__((ext_vector_type(16))) unsigned short  v16us;
typedef __attribute__((ext_vector_type(8)))  unsigned short  v8us;
typedef __attribute__((ext_vector_type(4)))  unsigned int    tdm_u4;
typedef __attribute__((ext_vector_type(8)))  int             tdm_i8;
typedef __attribute__((ext_vector_type(4)))  int             tdm_i4;

#if defined(__has_builtin)
#if __has_builtin(__builtin_amdgcn_tensor_load_to_lds) && \
    __has_builtin(__builtin_amdgcn_s_wait_tensorcnt)
#define HAVE_TDM 1
#endif
#endif

__device__ __forceinline__ unsigned short f2bf(float x) {
    unsigned int u = __float_as_uint(x);
    u += 0x7fffu + ((u >> 16) & 1u);           // round-to-nearest-even
    return (unsigned short)(u >> 16);
}

__device__ __forceinline__ v16bf mk16(v8us a, v8us b) {
    v16us t = __builtin_shufflevector(a, b, 0,1,2,3,4,5,6,7,8,9,10,11,12,13,14,15);
    return __builtin_bit_cast(v16bf, t);
}

__device__ __forceinline__ v8f wmma_bf16(v16bf a, v16bf b, v8f c) {
    // (neg_a, A, neg_b, B, c_mod, C, reuse_a, reuse_b)
    return __builtin_amdgcn_wmma_f32_16x16x32_bf16(false, a, false, b, (short)0, c, false, false);
}

// ---------------------------------------------------------------------------
// fp32 -> bf16 flat convert (x), 4-wide vectorized.
__global__ __launch_bounds__(256) void cvt_x_kernel(const float* __restrict__ in,
                                                    unsigned short* __restrict__ out,
                                                    int n4) {
    int i = blockIdx.x * 256 + threadIdx.x;
    const int stride = gridDim.x * 256;
    for (; i < n4; i += stride) {
        float4 v = ((const float4*)in)[i];
        ushort4 o;
        o.x = f2bf(v.x); o.y = f2bf(v.y); o.z = f2bf(v.z); o.w = f2bf(v.w);
        ((ushort4*)out)[i] = o;
    }
}

// fp32 W[e][f] (1024x1024) -> bf16 Wt[f][e] (transposed for WMMA B gathers).
__global__ __launch_bounds__(256) void cvt_wt_kernel(const float* __restrict__ W,
                                                     unsigned short* __restrict__ Wt) {
    int i = blockIdx.x * 256 + threadIdx.x;
    const int stride = gridDim.x * 256;
    for (; i < (1 << 20); i += stride) {
        const int e = i >> 10, f = i & 1023;
        Wt[((size_t)f << 10) + e] = f2bf(W[i]);
    }
}

// ---------------------------------------------------------------------------
// Generic 8192x1024x1024 GEMM: out = A(bf16) x Bt^T(bf16) + bias.
// Workgroup = 8 waves, tile M=128 x N=64; wave = 16-row strip x 4 WMMA tiles.
// mode 0: Q  -> bf16 [bh][s][64], scaled by 0.125 (attn scale folded in)
// mode 1: K  -> bf16 [bh][s][64]
// mode 2: V  -> bf16 transposed [bh][64][2048]
// mode 3: out-> fp32 [bs][1024]  (final projection)
__global__ __launch_bounds__(256) void gemm_bf16_kernel(const unsigned short* __restrict__ A,
                                                        const unsigned short* __restrict__ Bt,
                                                        const float* __restrict__ bias,
                                                        void* __restrict__ out, int mode) {
    __shared__ unsigned short Bs[64 * 32];                 // 4 KB weight tile
    const int tid  = threadIdx.x;
    const int wave = tid >> 5, lane = tid & 31;
    const int ln   = lane & 15, hi = lane >> 4;
    const int mbase = blockIdx.x * 128 + wave * 16;
    const int nbase = blockIdx.y * 64;
    const size_t arow = (size_t)(mbase + ln) * 1024;

    v8f acc[4] = {};
    for (int kc = 0; kc < 1024; kc += 32) {
#ifdef HAVE_TDM
        // Tensor Data Mover: DMA the 64x32 bf16 weight tile straight to LDS.
        // D# per CDNA5 ISA (2-D tensor, groups 2/3 zero).
        if (wave == 0) {
            const unsigned long long ga =
                (unsigned long long)(uintptr_t)(Bt + (size_t)nbase * 1024 + kc);
            const unsigned int ldsa = (unsigned int)(uintptr_t)&Bs[0]; // LDS byte addr
            tdm_u4 g0;
            g0.x = 1u;                                     // count=1 (valid descriptor)
            g0.y = ldsa;                                   // lds_addr
            g0.z = (unsigned int)ga;                       // global_addr[31:0]
            g0.w = (unsigned int)((ga >> 32) & 0x01ffffffu) | 0x80000000u; // addr[56:32]|type=2
            tdm_i8 g1;
            g1[0] = 1 << 16;                               // data_size=1 (2 bytes)
            g1[1] = 0x04000000;                            // tensor_dim0=1024 (bits 63:48)
            g1[2] = 0x04000000;                            // tensor_dim1=1024 (bits 111:80)
            g1[3] = 32 << 16;                              // tile_dim0=32 (bits 127:112)
            g1[4] = 64;                                    // tile_dim1=64, tile_dim2=0
            g1[5] = 1024;                                  // tensor_dim0_stride[31:0]
            g1[6] = 0x04000000;                            // dim1_stride[15:0]=1024 (hi half)
            g1[7] = 0;                                     // dim1_stride[47:16]=0
            tdm_i4 gz = (tdm_i4)0;
#if __clang_major__ >= 23
            tdm_i8 gz8 = (tdm_i8)0;
            __builtin_amdgcn_tensor_load_to_lds(g0, g1, gz, gz, gz8, 0);
#else
            __builtin_amdgcn_tensor_load_to_lds(g0, g1, gz, gz, 0);
#endif
            __builtin_amdgcn_s_wait_tensorcnt(0);
        }
        __syncthreads();
#else
        {   // cooperative, fully coalesced weight-tile stage (64 rows x 64B)
            const int trow = tid >> 2, tcol = (tid & 3) * 8;
            *(v8us*)&Bs[trow * 32 + tcol] =
                *(const v8us*)&Bt[(size_t)(nbase + trow) * 1024 + kc + tcol];
        }
        __syncthreads();
#endif
        // A operand (16x32 bf16): K = 16*hi + e
        const unsigned short* ap = A + arow + kc + 16 * hi;
        v16bf av = mk16(*(const v8us*)ap, *(const v8us*)(ap + 8));
#pragma unroll
        for (int nt = 0; nt < 4; ++nt) {
            // B operand (32x16 bf16): lane column n=ln, K = 16*hi + 0..15
            const unsigned short* bp = &Bs[(nt * 16 + ln) * 32 + 16 * hi];
            v16bf bv = mk16(*(const v8us*)bp, *(const v8us*)(bp + 8));
            acc[nt] = wmma_bf16(av, bv, acc[nt]);
        }
        __syncthreads();
    }

#pragma unroll
    for (int nt = 0; nt < 4; ++nt) {
        const int f = nbase + nt * 16 + ln;
        const float bias_v = bias[f];
#pragma unroll
        for (int r = 0; r < 8; ++r) {
            const int mrow = mbase + r + 8 * hi;            // C: VGPR r -> rows r / r+8
            float v = acc[nt][r] + bias_v;
            if (mode == 3) {
                ((float*)out)[(size_t)mrow * 1024 + f] = v;
            } else {
                const int bb = mrow >> 11, s = mrow & 2047;
                const int h = f >> 6, d = f & 63;
                const int bh = bb * 16 + h;
                if (mode == 2) {
                    ((unsigned short*)out)[((size_t)bh * 64 + d) * 2048 + s] = f2bf(v);
                } else {
                    if (mode == 0) v *= 0.125f;             // 1/sqrt(64), lossless in bf16
                    ((unsigned short*)out)[((size_t)bh * 2048 + s) * 64 + d] = f2bf(v);
                }
            }
        }
    }
}

// ---------------------------------------------------------------------------
// Fused scores + softmax + ctx. Grid (16 qblocks, 64 bh). WG = 8 waves, each
// wave owns a 16-query strip x all 2048 keys (no inter-wave coupling).
__global__ __launch_bounds__(256) void attn_softmax_ctx_kernel(
        const unsigned short* __restrict__ Q,    // [64][2048][64] bf16, pre-scaled
        const unsigned short* __restrict__ K,    // [64][2048][64] bf16
        const unsigned short* __restrict__ Vt,   // [64][64][2048] bf16 (transposed)
        float* __restrict__ attn,                // [64][2048][2048] fp32 (d_out)
        unsigned short* __restrict__ ctx) {      // [8192][1024] bf16
    __shared__ unsigned short Pt[8][16 * 32];    // per-wave P-tile transpose (8 KB)
    const int tid = threadIdx.x, wave = tid >> 5, lane = tid & 31;
    const int ln = lane & 15, hi = lane >> 4;
    const int bh = blockIdx.y;
    const int qbase = blockIdx.x * 128 + wave * 16;
    const unsigned short* Qr = Q + ((size_t)bh * 2048 + qbase) * 64;
    const unsigned short* Kr = K + (size_t)bh * 2048 * 64;
    const unsigned short* Vr = Vt + (size_t)bh * 64 * 2048;
    float* arow = attn + ((size_t)bh * 2048 + qbase) * 2048;

    // Q strip (16x64) as two WMMA A operands, resident in registers.
    v16bf qa0, qa1;
    {
        const unsigned short* p = Qr + (size_t)ln * 64 + 16 * hi;
        qa0 = mk16(*(const v8us*)p,        *(const v8us*)(p + 8));
        qa1 = mk16(*(const v8us*)(p + 32), *(const v8us*)(p + 40));
    }

    // ---- Pass A: scores = Q K^T (raw, scaled) + per-lane online (m, l) ----
    float m[8], l[8];
#pragma unroll
    for (int r = 0; r < 8; ++r) { m[r] = -3.402823466e38f; l[r] = 0.f; }

    for (int kt = 0; kt < 128; ++kt) {
        const int kb = kt * 16;
        const unsigned short* kp = Kr + (size_t)(kb + ln) * 64 + 16 * hi;
        __builtin_prefetch(kp + 1024, 0, 0);     // next key tile
        v16bf b0 = mk16(*(const v8us*)kp,        *(const v8us*)(kp + 8));
        v16bf b1 = mk16(*(const v8us*)(kp + 32), *(const v8us*)(kp + 40));
        v8f s = {};
        s = wmma_bf16(qa0, b0, s);
        s = wmma_bf16(qa1, b1, s);
#pragma unroll
        for (int r = 0; r < 8; ++r) {
            const float sv = s[r];
            arow[(size_t)(r + 8 * hi) * 2048 + kb + ln] = sv;   // raw score
            const float mo = m[r], mn = fmaxf(mo, sv);
            l[r] = l[r] * __expf(mo - mn) + __expf(sv - mn);
            m[r] = mn;
        }
    }

    // Exact merge of per-lane stats across each 16-lane half (rows r / r+8).
    float mm[8], rl[8];
#pragma unroll
    for (int r = 0; r < 8; ++r) {
        float mr = m[r];
        mr = fmaxf(mr, __shfl_xor(mr, 1, 16));
        mr = fmaxf(mr, __shfl_xor(mr, 2, 16));
        mr = fmaxf(mr, __shfl_xor(mr, 4, 16));
        mr = fmaxf(mr, __shfl_xor(mr, 8, 16));
        float lp = l[r] * __expf(m[r] - mr);
        lp += __shfl_xor(lp, 1, 16);
        lp += __shfl_xor(lp, 2, 16);
        lp += __shfl_xor(lp, 4, 16);
        lp += __shfl_xor(lp, 8, 16);
        mm[r] = mr; rl[r] = 1.0f / lp;
    }

    // ---- Pass B: p = exp(s-m)/l -> attn (final), and ctx += P x V --------
    v8f cacc[4] = {};
    unsigned short* pts = &Pt[wave][0];
    for (int kc = 0; kc < 64; ++kc) {            // 32 keys per iteration
#pragma unroll
        for (int sub = 0; sub < 2; ++sub) {
            const int kb = (kc * 2 + sub) * 16;
#pragma unroll
            for (int r = 0; r < 8; ++r) {
                const size_t off = (size_t)(r + 8 * hi) * 2048 + kb + ln;
                const float sv = arow[off];
                const float p = __expf(sv - mm[r]) * rl[r];
                arow[off] = p;                   // final softmax value
                pts[(r + 8 * hi) * 32 + sub * 16 + ln] = f2bf(p);  // C->A transpose
            }
        }
        __syncthreads();                         // (wave-private LDS; ordering fence)
        // P tile as WMMA A operand (16 queries x 32 keys)
        const unsigned short* gp = pts + ln * 32 + 16 * hi;
        v16bf pv = mk16(*(const v8us*)gp, *(const v8us*)(gp + 8));
        const int keyb = kc * 32 + 16 * hi;
        __builtin_prefetch(Vr + keyb + 32, 0, 0);
#pragma unroll
        for (int nt = 0; nt < 4; ++nt) {         // d = nt*16 + ln, keys contiguous
            const unsigned short* vp = Vr + (size_t)(nt * 16 + ln) * 2048 + keyb;
            v16bf bv = mk16(*(const v8us*)vp, *(const v8us*)(vp + 8));
            cacc[nt] = wmma_bf16(pv, bv, cacc[nt]);
        }
        __syncthreads();
    }

    // ctx -> [b][s][h*64+d] bf16 for the final projection GEMM.
    const int b = bh >> 4, h = bh & 15;
#pragma unroll
    for (int nt = 0; nt < 4; ++nt) {
#pragma unroll
        for (int r = 0; r < 8; ++r) {
            const int srow = qbase + r + 8 * hi;
            ctx[((size_t)(b * 2048 + srow)) * 1024 + h * 64 + nt * 16 + ln] =
                f2bf(cacc[nt][r]);
        }
    }
}

// ---------------------------------------------------------------------------
extern "C" void kernel_launch(void* const* d_in, const int* in_sizes, int n_in,
                              void* d_out, int out_size, void* d_ws, size_t ws_size,
                              hipStream_t stream) {
    const float* x  = (const float*)d_in[0];
    const float* Wq = (const float*)d_in[1];
    const float* bq = (const float*)d_in[2];
    const float* Wk = (const float*)d_in[3];
    const float* bk = (const float*)d_in[4];
    const float* Wv = (const float*)d_in[5];
    const float* bv = (const float*)d_in[6];
    const float* Wo = (const float*)d_in[7];
    const float* bo = (const float*)d_in[8];

    // Workspace layout (88 MB total)
    char* ws = (char*)d_ws;
    unsigned short* xb   = (unsigned short*)ws;  ws += (size_t)8192 * 1024 * 2;   // 16 MB
    unsigned short* Wqt  = (unsigned short*)ws;  ws += (size_t)1024 * 1024 * 2;   //  2 MB
    unsigned short* Wkt  = (unsigned short*)ws;  ws += (size_t)1024 * 1024 * 2;
    unsigned short* Wvt  = (unsigned short*)ws;  ws += (size_t)1024 * 1024 * 2;
    unsigned short* Wot  = (unsigned short*)ws;  ws += (size_t)1024 * 1024 * 2;
    unsigned short* Qb   = (unsigned short*)ws;  ws += (size_t)64 * 2048 * 64 * 2; // 16 MB
    unsigned short* Kb   = (unsigned short*)ws;  ws += (size_t)64 * 2048 * 64 * 2;
    unsigned short* Vtb  = (unsigned short*)ws;  ws += (size_t)64 * 64 * 2048 * 2;
    unsigned short* ctxb = (unsigned short*)ws;  ws += (size_t)8192 * 1024 * 2;

    float* outp  = (float*)d_out;                // [4][2048][1024]
    float* attnp = outp + (size_t)8388608;       // [4][16][2048][2048]

    // 1) precision conversion / weight transpose
    cvt_x_kernel<<<2048, 256, 0, stream>>>(x, xb, 8192 * 1024 / 4);
    cvt_wt_kernel<<<1024, 256, 0, stream>>>(Wq, Wqt);
    cvt_wt_kernel<<<1024, 256, 0, stream>>>(Wk, Wkt);
    cvt_wt_kernel<<<1024, 256, 0, stream>>>(Wv, Wvt);
    cvt_wt_kernel<<<1024, 256, 0, stream>>>(Wo, Wot);

    // 2) QKV projections (WMMA)
    dim3 gg(64, 16);
    gemm_bf16_kernel<<<gg, 256, 0, stream>>>(xb, Wqt, bq, (void*)Qb,  0);
    gemm_bf16_kernel<<<gg, 256, 0, stream>>>(xb, Wkt, bk, (void*)Kb,  1);
    gemm_bf16_kernel<<<gg, 256, 0, stream>>>(xb, Wvt, bv, (void*)Vtb, 2);

    // 3) fused scores + softmax + ctx (WMMA)
    attn_softmax_ctx_kernel<<<dim3(16, 64), 256, 0, stream>>>(Qb, Kb, Vtb, attnp, ctxb);

    // 4) output projection (WMMA) -> fp32
    gemm_bf16_kernel<<<gg, 256, 0, stream>>>(ctxb, Wot, bo, (void*)outp, 3);

    (void)in_sizes; (void)n_in; (void)out_size; (void)ws_size;
}
